// SampledGraphormer_80753975099635
// MI455X (gfx1250) — matrix operand order repaired and checked by reading, not compiled, not run on hardware
//
#include <hip/hip_runtime.h>

// ---------------------------------------------------------------------------
// CDNA5 (gfx1250) implementation: every GEMM-shaped op goes through
// v_wmma_f32_16x16x32_bf16 (bf16 A/B, f32 accumulate). LDS tiles are staged
// in WMMA per-lane fragment layout so fragment loads are 2x ds_load_b128.
// GEMM is specialized on FULL (exact-tile) shapes; flash-attention softmax
// row reductions use DPP16 row_ror butterflies (VALU) instead of ds_bpermute.
// ---------------------------------------------------------------------------

typedef __attribute__((ext_vector_type(16))) __bf16 v16bf;
typedef __attribute__((ext_vector_type(8)))  float  v8f;

namespace {
constexpr int kN   = 2048;
constexpr int kE   = 32768;
constexpr int kIN  = 512;
constexpr int kND  = 256;
constexpr int kCD  = 64;
constexpr int kH   = 8;
constexpr int kDH  = 32;
constexpr int kFF  = 1024;
constexpr int kNL  = 4;
constexpr int kL   = 5;
constexpr int kMAXDEG = 256;
constexpr int kNF  = 20;
constexpr int kSH  = 64;
constexpr int kKSS = 24;   // subspace width: 1 trivial + 20 used + 3 guard
constexpr int kTIT = 20;   // subspace iterations
}

// --------------------------- input index map -------------------------------
// Assumption: harness flattens setup_inputs() dict depth-first in insertion
// order; params leaves appear as consecutive d_in entries.
enum {
  IX_X = 0, IX_EATTR, IX_EIDX, IX_PATHS, IX_Y,
  IP_NP_W1, IP_NP_B1, IP_NP_G1, IP_NP_BN1, IP_NP_W2, IP_NP_B2,          // node_proj
  IP_EP_W1, IP_EP_B1, IP_EP_G1, IP_EP_BN1, IP_EP_W2, IP_EP_B2,          // edge_proj
  IP_G_A, IP_G_B, IP_G_C, IP_G_U, IP_G_V, IP_G_WG,                      // gcn
  IP_G_LXG, IP_G_LXB, IP_G_LEG, IP_G_LEB,
  IP_ZIN, IP_ZOUT, IP_S_WPHI, IP_S_WRHO, IP_SPD,                        // 27..31
  IP_LAYER0 = 32,                 // per layer (13): Wq,Wk,Wv,Wo,Wc,g1,b1,g2,b2,F1,fb1,F2,fb2
  IP_HEAD_W1 = IP_LAYER0 + 4 * 13, IP_HEAD_B1, IP_HEAD_W2, IP_HEAD_B2   // 84..87
};

// --------------------------- bf16 helpers ----------------------------------
__device__ __forceinline__ unsigned short f2bf(float f) {
  union { float f; unsigned u; } v; v.f = f;
  unsigned r = (v.u + 0x7FFFu + ((v.u >> 16) & 1u)) >> 16;
  return (unsigned short)r;
}
__device__ __forceinline__ unsigned pack2bf(float lo, float hi) {
  return (unsigned)f2bf(lo) | ((unsigned)f2bf(hi) << 16);
}
// 32B-aligned fragment load from LDS: lowers to 2x ds_load_b128.
__device__ __forceinline__ v16bf frag_ld(const unsigned short* p) {
  return *(const v16bf*)p;
}

// DPP16 row-rotate butterfly reductions over a 16-lane row (wave32 = 2 rows;
// rows stay independent, matching the C-matrix half-wave row split).
template <int CTRL>
__device__ __forceinline__ float dpp16f(float x) {
  union { float f; int i; } a, r;
  a.f = x;
  r.i = __builtin_amdgcn_update_dpp(a.i, a.i, CTRL, 0xF, 0xF, true);
  return r.f;
}
__device__ __forceinline__ float row_max16(float x) {
  x = fmaxf(x, dpp16f<0x128>(x));   // row_ror:8
  x = fmaxf(x, dpp16f<0x124>(x));   // row_ror:4
  x = fmaxf(x, dpp16f<0x122>(x));   // row_ror:2
  x = fmaxf(x, dpp16f<0x121>(x));   // row_ror:1
  return x;
}
__device__ __forceinline__ float row_sum16(float x) {
  x += dpp16f<0x128>(x);
  x += dpp16f<0x124>(x);
  x += dpp16f<0x122>(x);
  x += dpp16f<0x121>(x);
  return x;
}

// WMMA per-lane fragment layouts (contiguous k order used by this code, wave32), inverted for
// staging:
//  A (16M x 32K):  lane = row + (k&16 ? 16 : 0); j = k & 15
//  B (32K x 16N):  lane = n   + (k&16 ? 16 : 0); j = k & 15
// Even-k pairs of A map to (same lane, j, j+1) -> packed 32-bit stores.

// --------------------------- generic WMMA GEMM -----------------------------
// C[M x Nc] = epilogue(gather(A)[M x K] @ B[K x Nc])
enum { GF_ACCUM = 1, GF_RELU = 2, GF_GELU = 4, GF_SIGMOID = 8, GF_ABS = 16 };

template <bool FULL>
__global__ __launch_bounds__(256)
void gemm_wmma(const float* __restrict__ A, int lda, const int* __restrict__ Aidx,
               const float* __restrict__ B, int ldb,
               float* __restrict__ C, int ldc,
               const float* __restrict__ bias, const float* __restrict__ resid,
               int M, int Nc, int K, int flags) {
  // fragment-major tiles: [tile][lane][j]
  __shared__ __attribute__((aligned(32))) unsigned short AsF[4][32][16]; // 64Mx32K
  __shared__ __attribute__((aligned(32))) unsigned short BsF[4][32][16]; // 32Kx64N
  const int tid = threadIdx.x;
  const int m0 = blockIdx.y * 64, n0 = blockIdx.x * 64;
  const int w = tid >> 5, lane = tid & 31;
  const int wy = w & 3, wx = w >> 2;          // 4x2 wave grid, 16x32 C per wave
  v8f acc0 = {}, acc1 = {};

  for (int k0 = 0; k0 < K; k0 += 32) {
    if (FULL) {
      // A: 64 rows x 16 k-pairs, coalesced float2 reads, packed b32 stores
#pragma unroll
      for (int u = 0; u < 4; ++u) {
        int i = tid + u * 256;
        int r = i >> 4, c2 = (i & 15) << 1;
        int ar = Aidx ? Aidx[m0 + r] : (m0 + r);
        const float* ap = &A[(long)ar * lda + k0 + c2];
        int lane2 = (r & 15) + ((c2 & 16) ? 16 : 0);
        int j     = c2 & 15;
        *(unsigned*)&AsF[r >> 4][lane2][j] = pack2bf(ap[0], ap[1]);
      }
      // B: 32 k x 64 n, coalesced along n
#pragma unroll
      for (int u = 0; u < 8; ++u) {
        int i = tid + u * 256;
        int r = i >> 6, c = i & 63;
        float v = B[(long)(k0 + r) * ldb + (n0 + c)];
        int lane2 = (c & 15) + ((r & 16) ? 16 : 0);
        BsF[c >> 4][lane2][r & 15] = f2bf(v);
      }
    } else {
      for (int i = tid; i < 64 * 16; i += 256) {
        int r = i >> 4, c2 = (i & 15) << 1;
        int gr = m0 + r, gk = k0 + c2;
        float v0 = 0.f, v1 = 0.f;
        if (gr < M) {
          int ar = Aidx ? Aidx[gr] : gr;
          const float* ap = &A[(long)ar * lda + gk];
          if (gk < K)     v0 = ap[0];
          if (gk + 1 < K) v1 = ap[1];
        }
        int lane2 = (r & 15) + ((c2 & 16) ? 16 : 0);
        int j     = c2 & 15;
        *(unsigned*)&AsF[r >> 4][lane2][j] = pack2bf(v0, v1);
      }
      for (int i = tid; i < 32 * 64; i += 256) {
        int r = i >> 6, c = i & 63;
        int gk = k0 + r, gn = n0 + c;
        float v = 0.f;
        if (gk < K && gn < Nc) v = B[(long)gk * ldb + gn];
        int lane2 = (c & 15) + ((r & 16) ? 16 : 0);
        BsF[c >> 4][lane2][r & 15] = f2bf(v);
      }
    }
    // speculative prefetch of the next A K-panel (global_prefetch_b8)
    if (k0 + 32 < K) {
      int gr = m0 + (tid & 63);
      if (FULL || gr < M) {
        int ar = Aidx ? Aidx[gr] : gr;
        __builtin_prefetch(&A[(long)ar * lda + k0 + 32], 0, 0);
      }
    }
    __syncthreads();
    v16bf a  = frag_ld(&AsF[wy][lane][0]);
    v16bf b0 = frag_ld(&BsF[wx * 2][lane][0]);
    v16bf b1 = frag_ld(&BsF[wx * 2 + 1][lane][0]);
    acc0 = __builtin_amdgcn_wmma_f32_16x16x32_bf16(false, a, false, b0, (short)0, acc0, false, false);
    acc1 = __builtin_amdgcn_wmma_f32_16x16x32_bf16(false, a, false, b1, (short)0, acc1, false, false);
    __syncthreads();
  }

  const int hi = (lane >> 4) & 1, ln15 = lane & 15;
#pragma unroll
  for (int r = 0; r < 8; ++r) {
    int row = m0 + wy * 16 + r + 8 * hi;
    if (!FULL && row >= M) continue;
#pragma unroll
    for (int t = 0; t < 2; ++t) {
      int col = n0 + wx * 32 + ln15 + t * 16;
      if (!FULL && col >= Nc) continue;
      float x = t ? acc1[r] : acc0[r];
      long off = (long)row * ldc + col;
      if (flags & GF_ACCUM)   x += C[off];
      if (bias)               x += bias[col];
      if (flags & GF_RELU)         x = fmaxf(x, 0.f);
      else if (flags & GF_GELU)    x = 0.5f * x * (1.f + erff(x * 0.70710678118f));
      else if (flags & GF_SIGMOID) x = 1.f / (1.f + __expf(-x));
      else if (flags & GF_ABS)     x = fabsf(x);
      if (resid)              x += resid[off];
      C[off] = x;
    }
  }
}

// --------------------------- LayerNorm over rows ---------------------------
// t = in (+ add); y = LN(t)*g + b; if relu: y=max(y,0); out = (res? res+y : y)
__global__ __launch_bounds__(256)
void ln_rows(const float* __restrict__ in, const float* __restrict__ add,
             const float* __restrict__ g, const float* __restrict__ b,
             const float* __restrict__ res, float* __restrict__ out,
             int cols, int relu) {
  __shared__ float r1[256], r2[256];
  const int row = blockIdx.x, tid = threadIdx.x;
  float s = 0.f, s2 = 0.f;
  for (int c = tid; c < cols; c += 256) {
    float t = in[(long)row * cols + c] + (add ? add[(long)row * cols + c] : 0.f);
    s += t; s2 += t * t;
  }
  r1[tid] = s; r2[tid] = s2; __syncthreads();
  for (int st = 128; st > 0; st >>= 1) {
    if (tid < st) { r1[tid] += r1[tid + st]; r2[tid] += r2[tid + st]; }
    __syncthreads();
  }
  const float mean = r1[0] / cols;
  const float var  = r2[0] / cols - mean * mean;
  const float inv  = rsqrtf(var + 1e-5f);
  for (int c = tid; c < cols; c += 256) {
    long off = (long)row * cols + c;
    float t = in[off] + (add ? add[off] : 0.f);
    float y = (t - mean) * inv * g[c] + b[c];
    if (relu) y = fmaxf(y, 0.f);
    if (res) y += res[off];
    out[off] = y;
  }
}

// --------------------------- small elementwise kernels ---------------------
__global__ void zero_f(float* p, int n) { int i = blockIdx.x * 256 + threadIdx.x; if (i < n) p[i] = 0.f; }
__global__ void zero_i(int* p, int n)   { int i = blockIdx.x * 256 + threadIdx.x; if (i < n) p[i] = 0; }
__global__ void copy_f(const float* a, float* b, int n) { int i = blockIdx.x * 256 + threadIdx.x; if (i < n) b[i] = a[i]; }
__global__ void ewdiv(float* num, const float* den, int n) {
  int i = blockIdx.x * 256 + threadIdx.x;
  if (i < n) num[i] = num[i] / (den[i] + 1e-6f);
}

__global__ void deg_count(const int* src, const int* dst, int* indeg, int* outdeg, int e) {
  int i = blockIdx.x * 256 + threadIdx.x;
  if (i < e) { atomicAdd(&outdeg[src[i]], 1); atomicAdd(&indeg[dst[i]], 1); }
}
__global__ void adj_build(const int* src, const int* dst, float* A, int e, int n) {
  int i = blockIdx.x * 256 + threadIdx.x;
  if (i < e) {
    atomicAdd(&A[(long)src[i] * n + dst[i]], 1.f);
    atomicAdd(&A[(long)dst[i] * n + src[i]], 1.f);
  }
}
__global__ void row_sum(const float* A, float* out, int n) {
  __shared__ float red[256];
  int r = blockIdx.x;
  float s = 0.f;
  for (int c = threadIdx.x; c < n; c += 256) s += A[(long)r * n + c];
  red[threadIdx.x] = s; __syncthreads();
  for (int st = 128; st > 0; st >>= 1) {
    if (threadIdx.x < st) red[threadIdx.x] += red[threadIdx.x + st];
    __syncthreads();
  }
  if (threadIdx.x == 0) out[r] = red[0];
}
__global__ void dis_make(const float* deg, float* dis, int n) {
  int i = blockIdx.x * 256 + threadIdx.x;
  if (i < n) dis[i] = (deg[i] > 0.f) ? rsqrtf(fmaxf(deg[i], 1e-6f)) : 0.f;
}
__global__ void lap_build(float* A, const float* dis, int n) {
  long i = (long)blockIdx.x * 256 + threadIdx.x;
  if (i < (long)n * n) {
    int r = (int)(i / n), c = (int)(i % n);
    A[i] = ((r == c) ? 1.f : 0.f) - dis[r] * A[i] * dis[c];
  }
}
__global__ void v_init(float* V, int n) {
  int i = blockIdx.x * 256 + threadIdx.x;
  if (i < n * kKSS) {
    int r = i / kKSS, j = i % kKSS;
    if (j == 0) { V[i] = 1.f; return; }
    unsigned u = (unsigned)(r * 2654435761u) ^ (unsigned)((j + 1) * 0x9E3779B9u);
    u ^= u >> 16; u *= 0x85EBCA6Bu; u ^= u >> 13;
    V[i] = ((float)(u & 0xFFFFFFu) / 16777216.0f) * 2.f - 1.f;
  }
}
// modified Gram-Schmidt on Z = 2V - Y (single workgroup, 256 threads, n=2048)
__global__ __launch_bounds__(256)
void mgs_kernel(float* V, const float* Y) {
  __shared__ float red[256];
  const int tid = threadIdx.x;
  for (int j = 0; j < kKSS; ++j) {
    float zr[8];
#pragma unroll
    for (int i = 0; i < 8; ++i) {
      int r = tid + i * 256;
      zr[i] = 2.f * V[r * kKSS + j] - Y[r * kKSS + j];
    }
    for (int p = 0; p < j; ++p) {
      float part = 0.f;
#pragma unroll
      for (int i = 0; i < 8; ++i) part += zr[i] * V[(tid + i * 256) * kKSS + p];
      red[tid] = part; __syncthreads();
      for (int st = 128; st > 0; st >>= 1) {
        if (tid < st) red[tid] += red[tid + st];
        __syncthreads();
      }
      float d = red[0]; __syncthreads();
#pragma unroll
      for (int i = 0; i < 8; ++i) zr[i] -= d * V[(tid + i * 256) * kKSS + p];
    }
    float part = 0.f;
#pragma unroll
    for (int i = 0; i < 8; ++i) part += zr[i] * zr[i];
    red[tid] = part; __syncthreads();
    for (int st = 128; st > 0; st >>= 1) {
      if (tid < st) red[tid] += red[tid + st];
      __syncthreads();
    }
    float inv = rsqrtf(fmaxf(red[0], 1e-20f)); __syncthreads();
#pragma unroll
    for (int i = 0; i < 8; ++i) V[(tid + i * 256) * kKSS + j] = zr[i] * inv;
    __syncthreads();
  }
}

// x += z_in[clip(indeg)] + z_out[clip(outdeg)] + pe
__global__ void z_add(float* x, const int* indeg, const int* outdeg,
                      const float* zin, const float* zout, const float* pe) {
  long i = (long)blockIdx.x * 256 + threadIdx.x;
  if (i < (long)kN * kND) {
    int n = (int)(i / kND), c = (int)(i % kND);
    int di = indeg[n];  di = di < kMAXDEG ? di : kMAXDEG;
    int dn = outdeg[n]; dn = dn < kMAXDEG ? dn : kMAXDEG;
    x[i] += zin[(long)di * kND + c] + zout[(long)dn * kND + c] + pe[i];
  }
}

// gated edge message: sig = sigmoid(ehat @ Wg); atomically segment-sum
__global__ __launch_bounds__(256)
void edge_msg(const float* __restrict__ ehat, const float* __restrict__ Wg,
              const float* __restrict__ xV, const int* __restrict__ src,
              const int* __restrict__ dst, float* num, float* den) {
  __shared__ float er[kCD];
  const int e = blockIdx.x, tid = threadIdx.x;
  if (tid < kCD) er[tid] = ehat[(long)e * kCD + tid];
  __syncthreads();
  const int s = src[e], d = dst[e];
  float a = 0.f;
#pragma unroll 8
  for (int k = 0; k < kCD; ++k) a += er[k] * Wg[k * kND + tid];
  float sig = 1.f / (1.f + __expf(-a));
  atomicAdd(&num[(long)d * kND + tid], sig * xV[(long)s * kND + tid]);
  atomicAdd(&den[(long)d * kND + tid], sig);
}

// dots[e][l] = <e_row, Wc[l]>
__global__ void path_dots(const float* __restrict__ e, const float* __restrict__ Wc,
                          float* __restrict__ dots) {
  int i = blockIdx.x * 256 + threadIdx.x;
  if (i < kE * kL) {
    int ei = i / kL, l = i % kL;
    float s = 0.f;
#pragma unroll 8
    for (int k = 0; k < kCD; ++k) s += e[(long)ei * kCD + k] * Wc[l * kCD + k];
    dots[i] = s;
  }
}
// cb[n][m] = spd[plen] + sum_valid(dots)/max(plen,1)
__global__ void bias_plane(const int* __restrict__ paths, const float* __restrict__ dots,
                           const float* __restrict__ spd, float* __restrict__ cb) {
  long i = (long)blockIdx.x * 256 + threadIdx.x;
  if (i < (long)kN * kN) {
    const int* p = paths + i * kL;
    int cnt = 0; float s = 0.f;
#pragma unroll
    for (int l = 0; l < kL; ++l) {
      int pi = p[l];
      if (pi >= 0) { ++cnt; s += dots[(long)pi * kL + l]; }
    }
    int pc = cnt < kL ? cnt : kL;
    float denom = (float)(cnt > 1 ? cnt : 1);
    cb[i] = spd[pc] + s / denom;
  }
}

// --------------------------- flash attention -------------------------------
// grid (N/128, H), 256 threads = 8 waves; wave handles 16 query rows.
__global__ __launch_bounds__(256)
void flash_attn(const float* __restrict__ qg, const float* __restrict__ kg,
                const float* __restrict__ vg, const float* __restrict__ cb,
                float* __restrict__ og, int n) {
  const int h = blockIdx.y;
  const int tid = threadIdx.x, w = tid >> 5, lane = tid & 31;
  const int hi = (lane >> 4) & 1, ln15 = lane & 15;
  const int qrow0 = blockIdx.x * 128 + w * 16;
  const float scale = 0.17677669529663687f;   // 1/sqrt(DH)

  // fragment-major staging: [tile][lane][j]
  __shared__ __attribute__((aligned(32))) unsigned short QsF[8][32][16];
  __shared__ __attribute__((aligned(32))) unsigned short KsF[2][32][16]; // B^T frags
  __shared__ __attribute__((aligned(32))) unsigned short VsF[2][32][16]; // B frags
  __shared__ __attribute__((aligned(32))) unsigned short PsF[8][32][16];

  // Q tile (16 rows x 32 dh) -> A fragment layout, packed pair stores
  for (int i = lane; i < 16 * 16; i += 32) {
    int r = i >> 4, c2 = (i & 15) << 1;
    const float* qp = &qg[(long)(qrow0 + r) * kND + h * kDH + c2];
    int lane2 = r + ((c2 & 16) ? 16 : 0);
    int j     = c2 & 15;
    *(unsigned*)&QsF[w][lane2][j] = pack2bf(qp[0] * scale, qp[1] * scale);
  }
  __syncthreads();
  const v16bf qa = frag_ld(&QsF[w][lane][0]);

  v8f acc0 = {}, acc1 = {};
  float mrow[8], lrow[8];
#pragma unroll
  for (int r = 0; r < 8; ++r) { mrow[r] = -3.0e38f; lrow[r] = 0.f; }

  for (int t0 = 0; t0 < n; t0 += 32) {
    __syncthreads();
    // K chunk: B[k=dh][n=key] fragment (from row-major [key][dh] global),
    // even-dh pairs share a lane -> packed b32 stores.
    for (int i = tid; i < 32 * 16; i += 256) {
      int key = i >> 4, c2 = (i & 15) << 1;
      const float* kp = &kg[(long)(t0 + key) * kND + h * kDH + c2];
      int lane2 = (key & 15) + ((c2 & 16) ? 16 : 0);
      *(unsigned*)&KsF[key >> 4][lane2][c2 & 15] = pack2bf(kp[0], kp[1]);
    }
    // V chunk: B[k=key][n=dh] fragment
    for (int i = tid; i < 32 * 32; i += 256) {
      int key = i >> 5, dh = i & 31;
      float v = vg[(long)(t0 + key) * kND + h * kDH + dh];
      int lane2 = (dh & 15) + ((key & 16) ? 16 : 0);
      VsF[dh >> 4][lane2][key & 15] = f2bf(v);
    }
    __syncthreads();

    v16bf kb0 = frag_ld(&KsF[0][lane][0]);   // keys t0..t0+15
    v16bf kb1 = frag_ld(&KsF[1][lane][0]);   // keys t0+16..t0+31
    v8f s0 = {}, s1 = {};
    s0 = __builtin_amdgcn_wmma_f32_16x16x32_bf16(false, qa, false, kb0, (short)0, s0, false, false);
    s1 = __builtin_amdgcn_wmma_f32_16x16x32_bf16(false, qa, false, kb1, (short)0, s1, false, false);

    float p0[8], p1[8];
#pragma unroll
    for (int r = 0; r < 8; ++r) {
      int row = qrow0 + r + 8 * hi;
      float e0 = s0[r] + cb[(long)row * n + t0 + ln15];
      float e1 = s1[r] + cb[(long)row * n + t0 + 16 + ln15];
      // per-row reductions within each 16-lane DPP row (VALU, no LDS)
      float mt = row_max16(fmaxf(e0, e1));
      float mn = fmaxf(mrow[r], mt);
      float al = __expf(mrow[r] - mn);
      float q0 = __expf(e0 - mn), q1 = __expf(e1 - mn);
      float rs = row_sum16(q0 + q1);
      lrow[r] = lrow[r] * al + rs;
      mrow[r] = mn;
      acc0[r] *= al; acc1[r] *= al;
      p0[r] = q0; p1[r] = q1;
    }
    // stage P (C-layout -> A-fragment layout):
    // (row, k=col): lane = row + (k&16 ? 16:0); j = k & 15
#pragma unroll
    for (int r = 0; r < 8; ++r) {
      int row = r + 8 * hi;
      PsF[w][row][ln15]      = f2bf(p0[r]);   // k = ln15
      PsF[w][row + 16][ln15] = f2bf(p1[r]);   // k = 16 + ln15
    }
    __syncthreads();
    v16bf pa  = frag_ld(&PsF[w][lane][0]);
    v16bf vb0 = frag_ld(&VsF[0][lane][0]);   // dh cols 0..15
    v16bf vb1 = frag_ld(&VsF[1][lane][0]);   // dh cols 16..31
    acc0 = __builtin_amdgcn_wmma_f32_16x16x32_bf16(false, pa, false, vb0, (short)0, acc0, false, false);
    acc1 = __builtin_amdgcn_wmma_f32_16x16x32_bf16(false, pa, false, vb1, (short)0, acc1, false, false);
  }

#pragma unroll
  for (int r = 0; r < 8; ++r) {
    int row = qrow0 + r + 8 * hi;
    float inv = 1.f / lrow[r];
    og[(long)row * kND + h * kDH + ln15]      = acc0[r] * inv;
    og[(long)row * kND + h * kDH + 16 + ln15] = acc1[r] * inv;
  }
}

// --------------------------- head: pred = hid @ W2 + b2 --------------------
__global__ __launch_bounds__(256)
void head_pred(const float* __restrict__ hid, const float* __restrict__ W2,
               const float* __restrict__ b2, float* __restrict__ out, int e) {
  int wglobal = blockIdx.x * 8 + (threadIdx.x >> 5);
  int lane = threadIdx.x & 31;
  if (wglobal >= e) return;
  float s = 0.f;
  for (int c = lane; c < kND; c += 32) s += hid[(long)wglobal * kND + c] * W2[c];
#pragma unroll
  for (int msk = 16; msk > 0; msk >>= 1) s += __shfl_xor(s, msk, 32);
  if (lane == 0) out[wglobal] = s + b2[0];
}

// --------------------------- host orchestration ----------------------------
static inline void gemm(hipStream_t st, const float* A, int lda, const int* idx,
                        const float* B, int ldb, float* C, int ldc,
                        const float* bias, const float* resid,
                        int M, int Nc, int K, int flags) {
  dim3 g((Nc + 63) / 64, (M + 63) / 64);
  const bool full = (M % 64 == 0) && (Nc % 64 == 0) && (K % 32 == 0);
  if (full)
    gemm_wmma<true><<<g, 256, 0, st>>>(A, lda, idx, B, ldb, C, ldc, bias, resid, M, Nc, K, flags);
  else
    gemm_wmma<false><<<g, 256, 0, st>>>(A, lda, idx, B, ldb, C, ldc, bias, resid, M, Nc, K, flags);
}
static inline int ceildiv(long a, int b) { return (int)((a + b - 1) / b); }

extern "C" void kernel_launch(void* const* d_in, const int* in_sizes, int n_in,
                              void* d_out, int out_size, void* d_ws, size_t ws_size,
                              hipStream_t stream) {
  (void)in_sizes; (void)n_in; (void)out_size; (void)ws_size;
  auto F = [&](int i) -> const float* { return (const float*)d_in[i]; };
  const int* eidx  = (const int*)d_in[IX_EIDX];
  const int* src   = eidx;
  const int* dst   = eidx + kE;
  const int* paths = (const int*)d_in[IX_PATHS];
  float* out = (float*)d_out;

  // ---- workspace layout ----
  char* wsp = (char*)d_ws;
  size_t cur = 0;
  auto alloc = [&](size_t nbytes) -> void* {
    void* p = wsp + cur; cur += (nbytes + 255) & ~(size_t)255; return p;
  };
  float* xp   = (float*)alloc((size_t)kN * kND * 4);
  float* eb   = (float*)alloc((size_t)kE * kCD * 4);
  float* ehat = (float*)alloc((size_t)kE * kCD * 4);
  float* xV   = (float*)alloc((size_t)kN * kND * 4);
  float* num  = (float*)alloc((size_t)kN * kND * 4);
  float* den  = (float*)alloc((size_t)kN * kND * 4);
  float* tmpx = (float*)alloc((size_t)kN * kND * 4);
  float* qb   = (float*)alloc((size_t)kN * kND * 4);
  float* kb   = (float*)alloc((size_t)kN * kND * 4);
  float* vb   = (float*)alloc((size_t)kN * kND * 4);
  float* ob   = (float*)alloc((size_t)kN * kND * 4);
  float* buf1 = (float*)alloc((size_t)kE * kND * 4);   // largest scratch (E x 256)
  float* Amat = (float*)alloc((size_t)kN * kN * 4);
  float* cbuf = (float*)alloc((size_t)kN * kN * 4);
  float* Vss  = (float*)alloc((size_t)kN * kKSS * 4);
  float* Yss  = (float*)alloc((size_t)kN * kKSS * 4);
  float* ph   = (float*)alloc((size_t)kN * kSH * 4);
  float* pe   = (float*)alloc((size_t)kN * kND * 4);
  float* dots = (float*)alloc((size_t)kE * kL * 4);
  float* degv = (float*)alloc((size_t)kN * 4);
  float* disv = (float*)alloc((size_t)kN * 4);
  int*   indeg  = (int*)alloc((size_t)kN * 4);
  int*   outdeg = (int*)alloc((size_t)kN * 4);

  // ---- node / edge projections ----
  gemm(stream, F(IX_X), kIN, nullptr, F(IP_NP_W1), 2 * kND, buf1, 2 * kND,
       F(IP_NP_B1), nullptr, kN, 2 * kND, kIN, GF_GELU);
  ln_rows<<<kN, 256, 0, stream>>>(buf1, nullptr, F(IP_NP_G1), F(IP_NP_BN1), nullptr, buf1, 2 * kND, 0);
  gemm(stream, buf1, 2 * kND, nullptr, F(IP_NP_W2), kND, xp, kND,
       F(IP_NP_B2), nullptr, kN, kND, 2 * kND, 0);

  gemm(stream, F(IX_EATTR), 32, nullptr, F(IP_EP_W1), 2 * kCD, buf1, 2 * kCD,
       F(IP_EP_B1), nullptr, kE, 2 * kCD, 32, GF_GELU);
  ln_rows<<<kE, 256, 0, stream>>>(buf1, nullptr, F(IP_EP_G1), F(IP_EP_BN1), nullptr, buf1, 2 * kCD, 0);
  gemm(stream, buf1, 2 * kCD, nullptr, F(IP_EP_W2), kCD, eb, kCD,
       F(IP_EP_B2), nullptr, kE, kCD, 2 * kCD, 0);

  // ---- gated GCN layer ----
  gemm(stream, xp, kND, src, F(IP_G_A), kCD, ehat, kCD, nullptr, nullptr, kE, kCD, kND, 0);
  gemm(stream, xp, kND, dst, F(IP_G_B), kCD, ehat, kCD, nullptr, nullptr, kE, kCD, kND, GF_ACCUM);
  gemm(stream, eb, kCD, nullptr, F(IP_G_C), kCD, ehat, kCD, nullptr, nullptr, kE, kCD, kCD, GF_ACCUM);
  gemm(stream, xp, kND, nullptr, F(IP_G_V), kND, xV, kND, nullptr, nullptr, kN, kND, kND, 0);

  zero_f<<<ceildiv((long)kN * kND, 256), 256, 0, stream>>>(num, kN * kND);
  zero_f<<<ceildiv((long)kN * kND, 256), 256, 0, stream>>>(den, kN * kND);
  edge_msg<<<kE, 256, 0, stream>>>(ehat, F(IP_G_WG), xV, src, dst, num, den);
  ewdiv<<<ceildiv((long)kN * kND, 256), 256, 0, stream>>>(num, den, kN * kND);

  gemm(stream, xp, kND, nullptr, F(IP_G_U), kND, tmpx, kND, nullptr, nullptr, kN, kND, kND, 0);
  // x = x + relu(LN(x@U + num/den))
  ln_rows<<<kN, 256, 0, stream>>>(tmpx, num, F(IP_G_LXG), F(IP_G_LXB), xp, xp, kND, 1);
  // e = e + relu(LN(ehat))
  ln_rows<<<kE, 256, 0, stream>>>(ehat, nullptr, F(IP_G_LEG), F(IP_G_LEB), eb, eb, kCD, 1);

  // ---- degrees ----
  zero_i<<<ceildiv(kN, 256), 256, 0, stream>>>(indeg, kN);
  zero_i<<<ceildiv(kN, 256), 256, 0, stream>>>(outdeg, kN);
  deg_count<<<ceildiv(kE, 256), 256, 0, stream>>>(src, dst, indeg, outdeg, kE);

  // ---- Laplacian PE via WMMA subspace iteration (replaces eigh) ----
  zero_f<<<ceildiv((long)kN * kN, 256), 256, 0, stream>>>(Amat, kN * kN);
  adj_build<<<ceildiv(kE, 256), 256, 0, stream>>>(src, dst, Amat, kE, kN);
  row_sum<<<kN, 256, 0, stream>>>(Amat, degv, kN);
  dis_make<<<ceildiv(kN, 256), 256, 0, stream>>>(degv, disv, kN);
  lap_build<<<ceildiv((long)kN * kN, 256), 256, 0, stream>>>(Amat, disv, kN);
  v_init<<<ceildiv(kN * kKSS, 256), 256, 0, stream>>>(Vss, kN);
  for (int it = 0; it < kTIT; ++it) {
    gemm(stream, Amat, kN, nullptr, Vss, kKSS, Yss, kKSS, nullptr, nullptr, kN, kKSS, kN, 0);
    mgs_kernel<<<1, 256, 0, stream>>>(Vss, Yss);
  }
  // pe = |eig @ Wphi| @ Wrho  (relu(z)+relu(-z) == |z|), eig = Vss cols 1..20
  gemm(stream, Vss + 1, kKSS, nullptr, F(IP_S_WPHI), kSH, ph, kSH, nullptr, nullptr, kN, kSH, kNF, GF_ABS);
  gemm(stream, ph, kSH, nullptr, F(IP_S_WRHO), kND, pe, kND, nullptr, nullptr, kN, kND, kSH, 0);
  z_add<<<ceildiv((long)kN * kND, 256), 256, 0, stream>>>(xp, indeg, outdeg, F(IP_ZIN), F(IP_ZOUT), pe);

  // ---- transformer layers ----
  for (int l = 0; l < kNL; ++l) {
    const int base = IP_LAYER0 + l * 13;
    const float* Wq = F(base + 0); const float* Wk = F(base + 1);
    const float* Wv = F(base + 2); const float* Wo = F(base + 3);
    const float* Wc = F(base + 4);
    const float* g1 = F(base + 5); const float* b1 = F(base + 6);
    const float* g2 = F(base + 7); const float* b2 = F(base + 8);
    const float* F1 = F(base + 9); const float* fb1 = F(base + 10);
    const float* F2 = F(base + 11); const float* fb2 = F(base + 12);

    path_dots<<<ceildiv(kE * kL, 256), 256, 0, stream>>>(eb, Wc, dots);
    bias_plane<<<ceildiv((long)kN * kN, 256), 256, 0, stream>>>(paths, dots, F(IP_SPD), cbuf);

    gemm(stream, xp, kND, nullptr, Wq, kND, qb, kND, nullptr, nullptr, kN, kND, kND, 0);
    gemm(stream, xp, kND, nullptr, Wk, kND, kb, kND, nullptr, nullptr, kN, kND, kND, 0);
    gemm(stream, xp, kND, nullptr, Wv, kND, vb, kND, nullptr, nullptr, kN, kND, kND, 0);

    dim3 fg(kN / 128, kH);
    flash_attn<<<fg, 256, 0, stream>>>(qb, kb, vb, cbuf, ob, kN);

    gemm(stream, ob, kND, nullptr, Wo, kND, tmpx, kND, nullptr, nullptr, kN, kND, kND, 0);
    ln_rows<<<kN, 256, 0, stream>>>(xp, tmpx, g1, b1, nullptr, xp, kND, 0);   // x = LN(x + o@Wo)

    gemm(stream, xp, kND, nullptr, F1, kFF, buf1, kFF, fb1, nullptr, kN, kFF, kND, GF_RELU);
    gemm(stream, buf1, kFF, nullptr, F2, kND, tmpx, kND, fb2, nullptr, kN, kND, kFF, 0);
    ln_rows<<<kN, 256, 0, stream>>>(xp, tmpx, g2, b2, nullptr, xp, kND, 0);   // x = LN(x + f)
  }

  // ---- edge head: relu([x_src | x_dst | e] @ W1 + b1) @ W2 + b2 ----
  const float* W1 = F(IP_HEAD_W1);   // (576, 256) row-major
  gemm(stream, xp, kND, src, W1,                 kND, buf1, kND, F(IP_HEAD_B1), nullptr, kE, kND, kND, 0);
  gemm(stream, xp, kND, dst, W1 + 256 * kND,     kND, buf1, kND, nullptr,       nullptr, kE, kND, kND, GF_ACCUM);
  gemm(stream, eb, kCD, nullptr, W1 + 512 * kND, kND, buf1, kND, nullptr,       nullptr, kE, kND, kCD, GF_ACCUM | GF_RELU);
  head_pred<<<ceildiv(kE, 8), 256, 0, stream>>>(buf1, F(IP_HEAD_W2), F(IP_HEAD_B2), out, kE);

  // second tuple element: y passthrough
  copy_f<<<ceildiv(kE, 256), 256, 0, stream>>>(F(IX_Y), out + kE, kE);
}
